// NLSPNModel_Adapt_16106127360390
// MI455X (gfx1250) — matrix operand order repaired and checked by reading, not compiled, not run on hardware
//
#include <hip/hip_runtime.h>
#include <hip/hip_bf16.h>
#include <math.h>

typedef __attribute__((ext_vector_type(16))) _Float16 v16h;
typedef __attribute__((ext_vector_type(8)))  float    v8f;

#define BB  4
#define HH  240
#define WW  1216
#define HWW (HH * WW)
#define NN  (BB * HWW)

union V16H { v16h v; _Float16 e[16]; };
union V8F  { v8f  v; float    e[8];  };

// Bilinear sample with zeros outside borders (matches reference _bilinear)
__device__ __forceinline__ float bilin_zero(const float* __restrict__ img,
                                            float yy, float xx) {
  float y0f = floorf(yy), x0f = floorf(xx);
  float wy = yy - y0f,   wx = xx - x0f;
  int y0 = (int)y0f, x0 = (int)x0f;
  int y1 = y0 + 1,   x1 = x0 + 1;
  bool vy0 = (y0 >= 0) && (y0 < HH);
  bool vy1 = (y1 >= 0) && (y1 < HH);
  bool vx0 = (x0 >= 0) && (x0 < WW);
  bool vx1 = (x1 >= 0) && (x1 < WW);
  int yc0 = min(max(y0, 0), HH - 1), yc1 = min(max(y1, 0), HH - 1);
  int xc0 = min(max(x0, 0), WW - 1), xc1 = min(max(x1, 0), WW - 1);
  float f00 = (vy0 && vx0) ? img[yc0 * WW + xc0] : 0.f;
  float f01 = (vy0 && vx1) ? img[yc0 * WW + xc1] : 0.f;
  float f10 = (vy1 && vx0) ? img[yc1 * WW + xc0] : 0.f;
  float f11 = (vy1 && vx1) ? img[yc1 * WW + xc1] : 0.f;
  float omwy = 1.f - wy, omwx = 1.f - wx;
  return f00 * omwy * omwx + f01 * omwy * wx + f10 * wy * omwx + f11 * wy * wx;
}

// ---------------------------------------------------------------------------
// Kernel 1: 3x3 conv head (8ch -> 24ch) as implicit GEMM on WMMA.
// M = 16 pixels (contiguous x in one row), N = 24 outch (2 tiles of 16),
// K = 72 (ic*9 + ky*3 + kx) padded to 96 (3 chunks of 32).
// One wave per 16-pixel tile; 128-thread blocks = 64 pixels of one row.
// ---------------------------------------------------------------------------
__global__ __launch_bounds__(128) void conv_wmma_kernel(
    const float* __restrict__ g,      // guidance [B,8,H,W]
    const float* __restrict__ Wf,     // W_oa flat [24*72]
    float* __restrict__ convOut)      // [B,24,H,W]
{
  const int b     = blockIdx.z;
  const int y     = blockIdx.y;
  const int wave  = threadIdx.x >> 5;
  const int lane  = threadIdx.x & 31;
  const int xbase = blockIdx.x * 64 + wave * 16;
  const int m     = lane & 15;        // pixel row of A held by this lane
  const int hf    = lane >> 4;        // lane half selects K sub-blocks
  const int col   = lane & 15;        // output-channel column of B/D

  // --- A fragments: im2col rows of 16 pixels, f32 -> f16 ---
  V16H a[3];
#pragma unroll
  for (int q = 0; q < 3; ++q) {
#pragma unroll
    for (int i = 0; i < 16; ++i) {
      // 16-bit A 16x32 layout: lanes<16 hold K{0..7,16..23}, lanes>=16 K{8..15,24..31}
      int kl = (i < 8) ? (i + hf * 8) : (16 + (i - 8) + hf * 8);
      int K  = q * 32 + kl;
      float v = 0.f;
      if (K < 72) {
        int c  = K / 9;
        int r  = K - c * 9;
        int ky = r / 3;
        int kx = r - ky * 3;
        int yy = y + ky - 1;
        int xx = xbase + m + kx - 1;
        if (yy >= 0 && yy < HH && xx >= 0 && xx < WW)
          v = g[((b * 8 + c) * HH + yy) * WW + xx];
      }
      a[q].e[i] = (_Float16)v;
    }
  }

  // --- B fragments: weights, column = output channel ---
  V16H bf[2][3];
#pragma unroll
  for (int nt = 0; nt < 2; ++nt) {
    int oc = nt * 16 + col;
#pragma unroll
    for (int q = 0; q < 3; ++q) {
#pragma unroll
      for (int i = 0; i < 16; ++i) {
        // B 32x16 layout: lane = column N; same K order as A: lanes<16 K{0..7,16..23}, lanes>=16 K{8..15,24..31}
        int K = q * 32 + ((i < 8) ? (i + hf * 8) : (16 + (i - 8) + hf * 8));
        float w = (oc < 24 && K < 72) ? Wf[oc * 72 + K] : 0.f;
        bf[nt][q].e[i] = (_Float16)w;
      }
    }
  }

  // --- 6 chained WMMAs: 2 N-tiles x 3 K-chunks ---
  V8F acc0 = {}, acc1 = {};
#pragma unroll
  for (int q = 0; q < 3; ++q) {
    acc0.v = __builtin_amdgcn_wmma_f32_16x16x32_f16(
        false, a[q].v, false, bf[0][q].v, (short)0, acc0.v, false, false);
    acc1.v = __builtin_amdgcn_wmma_f32_16x16x32_f16(
        false, a[q].v, false, bf[1][q].v, (short)0, acc1.v, false, false);
  }

  // --- store D: VGPR r -> pixel (xbase + r + 8*half), lane -> channel ---
#pragma unroll
  for (int r = 0; r < 8; ++r) {
    int x = xbase + r + hf * 8;               // W % 64 == 0 -> always in range
    convOut[((b * 24 + col) * HH + y) * WW + x] = acc0.e[r];
    int oc1 = 16 + col;
    if (oc1 < 24)
      convOut[((b * 24 + oc1) * HH + y) * WW + x] = acc1.e[r];
  }
}

// ---------------------------------------------------------------------------
// Kernel 2: per-pixel affinity normalization + confidence modulation +
// loop-invariant sampling coordinates. Also initializes feat0 (blended).
// ---------------------------------------------------------------------------
__global__ __launch_bounds__(256) void affinity_kernel(
    const float* __restrict__ convOut,   // [B,24,H,W]
    const float* __restrict__ bias,      // [24]
    const float* __restrict__ conf,      // [B,1,H,W]
    const float* __restrict__ feat_init, // [B,1,H,W]
    const float* __restrict__ feat_fix,  // [B,1,H,W]
    const float* __restrict__ aff_scale_p,
    float* __restrict__ aff,             // [9][B*H*W]
    float* __restrict__ ys,              // [9][B*H*W]
    float* __restrict__ xs,              // [9][B*H*W]
    float* __restrict__ feat0)           // blended initial field
{
  int p = blockIdx.x * blockDim.x + threadIdx.x;
  if (p >= NN) return;
  int b   = p / HWW;
  int rem = p - b * HWW;
  int y   = rem / WW;
  int x   = rem - y * WW;

  float cv[24];
#pragma unroll
  for (int ch = 0; ch < 24; ++ch)
    cv[ch] = convOut[(b * 24 + ch) * HWW + rem] + bias[ch];

  float scale = aff_scale_p[0] + 1e-8f;
  const float* cimg = conf + b * HWW;

  // TGASS affinity with confidence modulation
  float av[8];
  float abssum = 0.f;
#pragma unroll
  for (int k = 0; k < 8; ++k) {
    float dy = cv[k];        // o1 = dy
    float dx = cv[8 + k];    // o2 = dx
    float ca = bilin_zero(cimg, (float)y + dy, (float)x + dx);
    float aa = tanhf(cv[16 + k]) / scale * ca;
    av[k] = aa;
    abssum += fabsf(aa);
  }
  float s = fmaxf(abssum + 1e-4f, 1.f);
  float inv = 1.f / s;
  float asum = 0.f;
#pragma unroll
  for (int k = 0; k < 8; ++k) { av[k] *= inv; asum += av[k]; }
  float aref = 1.f - asum;

#pragma unroll
  for (int j = 0; j < 9; ++j) {
    float dy, dx, aj;
    if (j < 4)       { dy = cv[j];     dx = cv[8 + j];     aj = av[j];     }
    else if (j == 4) { dy = 0.f;       dx = 0.f;           aj = aref;      }
    else             { dy = cv[j - 1]; dx = cv[8 + j - 1]; aj = av[j - 1]; }
    int kh = j / 3 - 1;
    int kw = j % 3 - 1;
    aff[j * NN + p] = aj;
    ys[j * NN + p]  = (float)(y + kh) + dy;
    xs[j * NN + p]  = (float)(x + kw) + dx;
  }

  // feat entering sweep 0 is already blended with the sparse fixed input
  float fx = feat_fix[p];
  feat0[p] = (fx > 0.f) ? fx : feat_init[p];
}

// ---------------------------------------------------------------------------
// Kernel 3: one propagation sweep. Input field is pre-blended; output is
// written blended for the next sweep, or raw on the last sweep.
// ---------------------------------------------------------------------------
__global__ __launch_bounds__(256) void prop_kernel(
    const float* __restrict__ featIn,
    const float* __restrict__ feat_fix,
    const float* __restrict__ aff,
    const float* __restrict__ ys,
    const float* __restrict__ xs,
    float* __restrict__ featOut,
    int last)
{
  int p = blockIdx.x * blockDim.x + threadIdx.x;
  if (p >= NN) return;
  int b = p / HWW;
  const float* fimg = featIn + b * HWW;

  float acc = 0.f;
#pragma unroll
  for (int j = 0; j < 9; ++j) {
    float aj = aff[j * NN + p];
    float v  = bilin_zero(fimg, ys[j * NN + p], xs[j * NN + p]);
    acc = fmaf(aj, v, acc);
  }
  if (last) {
    featOut[p] = acc;                       // reference returns unblended
  } else {
    float fx = feat_fix[p];
    featOut[p] = (fx > 0.f) ? fx : acc;     // pre-blend for next sweep
  }
}

// ---------------------------------------------------------------------------
extern "C" void kernel_launch(void* const* d_in, const int* in_sizes, int n_in,
                              void* d_out, int out_size, void* d_ws, size_t ws_size,
                              hipStream_t stream) {
  const float* feat_init  = (const float*)d_in[0];
  const float* guidance   = (const float*)d_in[1];
  const float* confidence = (const float*)d_in[2];
  const float* feat_fix   = (const float*)d_in[3];
  const float* W_oa       = (const float*)d_in[4];
  const float* b_oa       = (const float*)d_in[5];
  const float* aff_scale  = (const float*)d_in[6];
  float* out = (float*)d_out;

  // workspace layout (all f32): conv 24N | aff 9N | ys 9N | xs 9N | featA N | featB N
  float* convOut = (float*)d_ws;
  float* affb    = convOut + (size_t)24 * NN;
  float* ysb     = affb    + (size_t)9 * NN;
  float* xsb     = ysb     + (size_t)9 * NN;
  float* featA   = xsb     + (size_t)9 * NN;
  float* featB   = featA   + (size_t)NN;

  dim3 gridC(WW / 64, HH, BB);
  conv_wmma_kernel<<<gridC, 128, 0, stream>>>(guidance, W_oa, convOut);

  int blocks = (NN + 255) / 256;
  affinity_kernel<<<blocks, 256, 0, stream>>>(
      convOut, b_oa, confidence, feat_init, feat_fix, aff_scale,
      affb, ysb, xsb, featA);

  float* bufs[2] = {featA, featB};
  const float* cur = featA;
  for (int t = 0; t < 18; ++t) {
    float* o = (t == 17) ? out : bufs[(t + 1) & 1];
    prop_kernel<<<blocks, 256, 0, stream>>>(
        cur, feat_fix, affb, ysb, xsb, o, (t == 17) ? 1 : 0);
    cur = o;
  }
}
